// InfiniAttention_85959475462671
// MI455X (gfx1250) — matrix operand order repaired and checked by reading, not compiled, not run on hardware
//
#include <hip/hip_runtime.h>

// InfiniAttention for MI455X (gfx1250, wave32, WMMA).
// Precision: bf16 inputs to all matmuls, f32 accumulation/state (v_wmma_f32_16x16x32_bf16).
// Shapes hardcoded to the reference: B=4, L=4096, D=1024, H=16, dh=64, C=8, S=512.
// All LDS operand tiles are stored so that WMMA fragment reads are contiguous
// (2x ds_load_b128 per fragment) -- transposed staging for B-side operands.
// Pure memory->LDS copies use the gfx1250 async-to-LDS path when available.

typedef __attribute__((ext_vector_type(16))) __bf16 v16bf;
typedef __attribute__((ext_vector_type(8)))  float  v8f;
typedef int i32x4 __attribute__((vector_size(16)));

union BFrag { v16bf v; __bf16 h[16]; };
union FFrag { v8f  v; float  f[8];  };

#if defined(__has_builtin)
#if __has_builtin(__builtin_amdgcn_global_load_async_to_lds_b128) && \
    __has_builtin(__builtin_amdgcn_s_wait_asynccnt)
#define USE_ASYNC_LDS 1
#endif
#endif

#ifdef USE_ASYNC_LDS
__device__ __forceinline__ void async_copy_b128(const void* gsrc, void* lds) {
  // builtin expects int4-vector pointers (per clang diagnostic probing)
  __builtin_amdgcn_global_load_async_to_lds_b128((i32x4*)gsrc, (i32x4*)lds, 0, 0);
}
__device__ __forceinline__ void async_wait0() {
  __builtin_amdgcn_s_wait_asynccnt(0);
}
#endif

__device__ __forceinline__ v8f wmma_bf16(v16bf a, v16bf b, v8f c) {
  // (neg_a, A, neg_b, B, c_mod, C, reuse_a, reuse_b)
  return __builtin_amdgcn_wmma_f32_16x16x32_bf16(false, a, false, b, (short)0, c, false, false);
}

__device__ __forceinline__ float sigma_f(float x) {   // ELU(alpha=1)
  return x >= 0.f ? x : (__expf(x) - 1.f);
}

// ---------------------------------------------------------------------------
// f32 -> bf16 (RNE via hardware convert)
// ---------------------------------------------------------------------------
__global__ __launch_bounds__(256) void cvt_f32_bf16(const float* __restrict__ src,
                                                    __bf16* __restrict__ dst, long n) {
  long i = (long)blockIdx.x * blockDim.x + threadIdx.x;
  if (i < n) dst[i] = (__bf16)src[i];
}

// ---------------------------------------------------------------------------
// C[N,Dn] = A[N,K](bf16) * W[K,Dn](bf16) + bias  (f32 out)
// 128x128x32 tile, 8 waves (4M x 2N), each wave 32x64 => 2x4 wmma frags.
// W tile staged TRANSPOSED in LDS so B fragments are contiguous b128 reads.
// A tile staged via async global->LDS (ASYNCcnt) when the builtin exists.
// ---------------------------------------------------------------------------
#define GM_BM 128
#define GM_BN 128
#define GM_BK 32

__global__ __launch_bounds__(256) void gemm_bf16_f32(
    const __bf16* __restrict__ A, const __bf16* __restrict__ W,
    const float* __restrict__ bias, float* __restrict__ C,
    int N, int K, int Dn) {
  __shared__ __bf16 As[GM_BM][GM_BK];   // 8 KB, row-major (A-fragment friendly)
  __shared__ __bf16 Bs[GM_BN][GM_BK];   // 8 KB, TRANSPOSED: Bs[n][k]

  const int tid  = threadIdx.x;
  const int lane = tid & 31, wave = tid >> 5;
  const int half = lane >> 4, ln = lane & 15;
  const int bm = blockIdx.y * GM_BM, bn = blockIdx.x * GM_BN;
  const int wm0 = (wave >> 1) * 32;    // wave row offset within tile
  const int wn0 = (wave & 1) * 64;     // wave col offset within tile

  FFrag acc[2][4];
  #pragma unroll
  for (int mt = 0; mt < 2; ++mt)
    #pragma unroll
    for (int nt = 0; nt < 4; ++nt)
      #pragma unroll
      for (int i = 0; i < 8; ++i) acc[mt][nt].f[i] = 0.f;

  const int ar = tid >> 1, ac = (tid & 1) * 16;   // A-tile load map (16 elems/thread)
  const int br = tid >> 3, bc = (tid & 7) * 16;   // B-tile load map (16 elems/thread)

  for (int kk = 0; kk < K; kk += GM_BK) {
    // A tile: coalesced global read -> LDS (async path bypasses VGPRs)
    const uint4* srcA = reinterpret_cast<const uint4*>(A + (long)(bm + ar) * K + kk + ac);
    uint4* dstA = reinterpret_cast<uint4*>(&As[ar][ac]);
#ifdef USE_ASYNC_LDS
    async_copy_b128(srcA + 0, dstA + 0);
    async_copy_b128(srcA + 1, dstA + 1);
#else
    dstA[0] = srcA[0]; dstA[1] = srcA[1];
#endif
    // W tile: coalesced global read, transposed LDS scatter (register path)
    const uint4* srcB = reinterpret_cast<const uint4*>(W + (long)(kk + br) * Dn + bn + bc);
    BFrag wrow;
    ((uint4*)wrow.h)[0] = srcB[0];
    ((uint4*)wrow.h)[1] = srcB[1];
    #pragma unroll
    for (int j = 0; j < 16; ++j) Bs[bc + j][br] = wrow.h[j];
    // hint next K tile into cache (global_prefetch_b8)
    __builtin_prefetch(A + (long)(bm + ar) * K + kk + GM_BK + ac, 0, 3);
    __builtin_prefetch(W + (long)(kk + GM_BK + br) * Dn + bn + bc, 0, 3);
#ifdef USE_ASYNC_LDS
    async_wait0();
#endif
    __syncthreads();

    // A fragments: 16x32 bf16, A-layout (lane<16: K 0-7/16-23; lane>=16: K 8-15/24-31)
    BFrag af[2];
    #pragma unroll
    for (int mt = 0; mt < 2; ++mt) {
      const int m = wm0 + mt * 16 + ln;
      ((uint4*)af[mt].h)[0] = *reinterpret_cast<const uint4*>(&As[m][half * 8]);
      ((uint4*)af[mt].h)[1] = *reinterpret_cast<const uint4*>(&As[m][16 + half * 8]);
    }
    #pragma unroll
    for (int nt = 0; nt < 4; ++nt) {
      BFrag bf;   // B-layout: lane n, K=half*8+j / 16+half*8+j from row Bs[n][*]
      const int n = wn0 + nt * 16 + ln;
      ((uint4*)bf.h)[0] = *reinterpret_cast<const uint4*>(&Bs[n][half * 8]);
      ((uint4*)bf.h)[1] = *reinterpret_cast<const uint4*>(&Bs[n][16 + half * 8]);
      #pragma unroll
      for (int mt = 0; mt < 2; ++mt)
        acc[mt][nt].v = wmma_bf16(af[mt].v, bf.v, acc[mt][nt].v);
    }
    __syncthreads();
  }

  #pragma unroll
  for (int mt = 0; mt < 2; ++mt)
    #pragma unroll
    for (int nt = 0; nt < 4; ++nt) {
      const int n = bn + wn0 + nt * 16 + ln;
      const float bv = bias ? bias[n] : 0.f;
      #pragma unroll
      for (int r = 0; r < 8; ++r) {
        const int m = bm + wm0 + mt * 16 + r + half * 8;
        C[(long)m * Dn + n] = acc[mt][nt].f[r] + bv;
      }
    }
}

// ---------------------------------------------------------------------------
// InfiniAttention core: one block per (b,h); sequential scan over 8 segments.
// ---------------------------------------------------------------------------
__global__ __launch_bounds__(256) void infini_attn(
    const float* __restrict__ Qf, const float* __restrict__ Kf,
    const float* __restrict__ Vf, float* __restrict__ O,
    __bf16* __restrict__ skbuf, __bf16* __restrict__ valbuf,
    const float* __restrict__ pbeta, const int* __restrict__ pmask,
    const int* __restrict__ pdelta) {
  constexpr int L = 4096, D = 1024, H = 16, DH = 64, S = 512, CSEG = 8;

  __shared__ float  Msh[DH][DH];        // compressive memory (f32), 16 KB
  __shared__ __bf16 MbT[DH][DH];        // bf16 M^T snapshot (B-frag friendly), 8 KB
  __shared__ float  zsh[DH];            // z vector
  __shared__ __bf16 Kch[32][DH];        // streamed K chunk, row-major (QK^T frags)
  __shared__ __bf16 VchT[DH][32];       // streamed V chunk TRANSPOSED (PV frags)
  __shared__ __bf16 SKchT[DH][32];      // sigma(k) chunk TRANSPOSED (M update)
  __shared__ __bf16 VLch[32][DH];       // val chunk row-major (M update)
  __shared__ __bf16 Pscr[8][16][32];    // per-wave P layout round-trip, 8 KB
  __shared__ float  rsq[S], rsk[S];     // row sums of sigma(q), sigma(k)
  __shared__ float  smax[S], ssum[S];   // online softmax state

  const int tid = threadIdx.x, lane = tid & 31, wave = tid >> 5;
  const int half = lane >> 4, ln = lane & 15;
  const int b = blockIdx.x >> 4, h = blockIdx.x & 15;
  const float g = 1.f / (1.f + __expf(-pbeta[0]));
  const int um = pmask[0], dr = pdelta[0];
  const float scale = 0.125f;  // 1/sqrt(64)

  __bf16* skb = skbuf  + (long)blockIdx.x * S * DH;
  __bf16* vlb = valbuf + (long)blockIdx.x * S * DH;

  for (int i = tid; i < DH * DH; i += 256) (&Msh[0][0])[i] = 0.f;
  if (tid < DH) zsh[tid] = 0.f;

  const int srow0 = wave * 64;          // this wave owns rows [srow0, srow0+64)

  BFrag qf[4][2];                       // raw q fragments (A-layout), kept across phases
  FFrag dacc[4][4];                     // attention PV accumulators

  for (int c = 0; c < CSEG; ++c) {
    smax[tid] = -1e30f; smax[tid + 256] = -1e30f;
    ssum[tid] = 0.f;    ssum[tid + 256] = 0.f;
    // bf16 snapshot of M^T for this segment's readouts (pre-update M)
    {
      const int e = tid >> 2, d0b = (tid & 3) * 16;
      #pragma unroll
      for (int j = 0; j < 16; ++j) MbT[e][d0b + j] = (__bf16)Msh[d0b + j][e];
    }
    __syncthreads();

    const long segbase = ((long)b * L + (long)c * S) * D + (long)h * DH;

    // ---- B-fragments of PRE-update M: contiguous b128 reads from MbT ----
    BFrag MB[2][4];
    #pragma unroll
    for (int kc = 0; kc < 2; ++kc)
      #pragma unroll
      for (int nt = 0; nt < 4; ++nt) {
        const int e = nt * 16 + ln;
        ((uint4*)MB[kc][nt].h)[0] = *(const uint4*)&MbT[e][kc * 32 + half * 8];
        ((uint4*)MB[kc][nt].h)[1] = *(const uint4*)&MbT[e][kc * 32 + 16 + half * 8];
      }

    // ================= Phase A: per-wave memory readout =================
    for (int rt = 0; rt < 4; ++rt) {
      const int r0 = srow0 + rt * 16;
      const int row = r0 + ln;
      const float* qp = Qf + segbase + (long)row * D;
      const float* kp = Kf + segbase + (long)row * D;

      BFrag sqf[2], skf[2];
      float rq = 0.f, rk = 0.f;
      #pragma unroll
      for (int kc = 0; kc < 2; ++kc) {
        #pragma unroll
        for (int j = 0; j < 8; ++j) {
          const float q0 = qp[kc * 32 + half * 8 + j];
          const float q1 = qp[kc * 32 + 16 + half * 8 + j];
          qf[rt][kc].h[j]     = (__bf16)q0;
          qf[rt][kc].h[8 + j] = (__bf16)q1;
          const float s0 = sigma_f(q0), s1 = sigma_f(q1);
          sqf[kc].h[j] = (__bf16)s0; sqf[kc].h[8 + j] = (__bf16)s1;
          rq += s0 + s1;
          const float k0 = kp[kc * 32 + half * 8 + j];
          const float k1 = kp[kc * 32 + 16 + half * 8 + j];
          const float t0 = sigma_f(k0), t1 = sigma_f(k1);
          skf[kc].h[j] = (__bf16)t0; skf[kc].h[8 + j] = (__bf16)t1;
          rk += t0 + t1;
        }
      }
      rq += __shfl_xor(rq, 16, 32);
      rk += __shfl_xor(rk, 16, 32);
      rsq[row] = rq; rsk[row] = rk;
      __builtin_amdgcn_wave_barrier();

      // stash sigma(k) rows (row-major bf16) for the M update
      #pragma unroll
      for (int kc = 0; kc < 2; ++kc) {
        *(uint4*)&skb[(long)row * DH + kc * 32 + half * 8]      = ((const uint4*)skf[kc].h)[0];
        *(uint4*)&skb[(long)row * DH + kc * 32 + 16 + half * 8] = ((const uint4*)skf[kc].h)[1];
      }

      // readouts: A = (sq@M)/(rowsum(sq)*z + eps); val = v - (sk@M)/(rowsum(sk)*z + eps)
      #pragma unroll
      for (int nt = 0; nt < 4; ++nt) {
        FFrag an, dn;
        #pragma unroll
        for (int i = 0; i < 8; ++i) { an.f[i] = 0.f; dn.f[i] = 0.f; }
        an.v = wmma_bf16(sqf[0].v, MB[0][nt].v, an.v);
        an.v = wmma_bf16(sqf[1].v, MB[1][nt].v, an.v);
        dn.v = wmma_bf16(skf[0].v, MB[0][nt].v, dn.v);
        dn.v = wmma_bf16(skf[1].v, MB[1][nt].v, dn.v);
        const int e = nt * 16 + ln;
        #pragma unroll
        for (int r = 0; r < 8; ++r) {
          const int srow = r0 + r + half * 8;
          const float vv = Vf[segbase + (long)srow * D + e];
          const float Aro = an.f[r] / (rsq[srow] * zsh[e] + 1e-6f);
          const float val = dr ? (vv - dn.f[r] / (rsk[srow] * zsh[e] + 1e-6f)) : vv;
          vlb[(long)srow * DH + e] = (__bf16)val;
          // reference's [C,B,H,S,dh] -> [B,D,L] -> [B,L,D] scatter
          const long f = (long)c * (H * S * DH) + (long)h * (S * DH) + (long)srow * DH + e;
          const long l = f & (L - 1), dd = f >> 12;
          O[((long)b * L + l) * D + dd] = g * Aro;   // dot part added in Phase B
        }
      }
      #pragma unroll
      for (int nt = 0; nt < 4; ++nt)
        #pragma unroll
        for (int i = 0; i < 8; ++i) dacc[rt][nt].f[i] = 0.f;
    }
    __threadfence_block();
    __syncthreads();

    // ================= Phase B: local softmax attention (flash-style) =================
    for (int tc = 0; tc < 16; ++tc) {
      __syncthreads();
      {
        const int e = tid * 8;
        const int trow = e >> 6, tcol = e & 63;
        const float* kr = Kf + segbase + (long)(tc * 32 + trow) * D + tcol;
        const float* vr = Vf + segbase + (long)(tc * 32 + trow) * D + tcol;
        #pragma unroll
        for (int j = 0; j < 8; ++j) {
          Kch[trow][tcol + j]  = (__bf16)kr[j];  // row-major for QK^T fragments
          VchT[tcol + j][trow] = (__bf16)vr[j];  // transposed for PV fragments
        }
      }
      __syncthreads();

      // K^T fragments (contiguous K rows) and V B-fragments (contiguous VchT rows)
      BFrag KB[2][2], VB[4];
      #pragma unroll
      for (int kc = 0; kc < 2; ++kc)
        #pragma unroll
        for (int sub = 0; sub < 2; ++sub) {
          const uint4* p = (const uint4*)&Kch[sub * 16 + ln][kc * 32 + half * 8];
          ((uint4*)KB[kc][sub].h)[0] = p[0];
          ((uint4*)KB[kc][sub].h)[1] = p[2];
        }
      #pragma unroll
      for (int nt = 0; nt < 4; ++nt) {
        const int n = nt * 16 + ln;
        ((uint4*)VB[nt].h)[0] = *(const uint4*)&VchT[n][half * 8];
        ((uint4*)VB[nt].h)[1] = *(const uint4*)&VchT[n][16 + half * 8];
      }

      for (int rt = 0; rt < 4; ++rt) {
        const int r0 = srow0 + rt * 16;
        FFrag lg[2];
        #pragma unroll
        for (int sub = 0; sub < 2; ++sub) {
          #pragma unroll
          for (int i = 0; i < 8; ++i) lg[sub].f[i] = 0.f;
          lg[sub].v = wmma_bf16(qf[rt][0].v, KB[0][sub].v, lg[sub].v);
          lg[sub].v = wmma_bf16(qf[rt][1].v, KB[1][sub].v, lg[sub].v);
        }
        float mx[8];
        #pragma unroll
        for (int r = 0; r < 8; ++r) {
          const int srow = r0 + r + half * 8;
          #pragma unroll
          for (int sub = 0; sub < 2; ++sub) {
            const int t = tc * 32 + sub * 16 + ln;
            float v = lg[sub].f[r] * scale;
            if (um && t > srow) v += 1e-9f;   // faithful to reference's (1-tril)*1e-9
            lg[sub].f[r] = v;
          }
          float m = fmaxf(lg[0].f[r], lg[1].f[r]);
          m = fmaxf(m, __shfl_xor(m, 1, 32));
          m = fmaxf(m, __shfl_xor(m, 2, 32));
          m = fmaxf(m, __shfl_xor(m, 4, 32));
          m = fmaxf(m, __shfl_xor(m, 8, 32));
          mx[r] = m;
        }
        #pragma unroll
        for (int r = 0; r < 8; ++r) {
          const int srow = r0 + r + half * 8;
          const float mo = smax[srow];
          const float nm = fmaxf(mo, mx[r]);
          const float corr = __expf(mo - nm);
          float ps = 0.f;
          #pragma unroll
          for (int sub = 0; sub < 2; ++sub) {
            const float p = __expf(lg[sub].f[r] - nm);
            ps += p;
            Pscr[wave][r + half * 8][sub * 16 + ln] = (__bf16)p;
          }
          ps += __shfl_xor(ps, 1, 32);
          ps += __shfl_xor(ps, 2, 32);
          ps += __shfl_xor(ps, 4, 32);
          ps += __shfl_xor(ps, 8, 32);
          smax[srow] = nm;
          ssum[srow] = ssum[srow] * corr + ps;
          #pragma unroll
          for (int nt = 0; nt < 4; ++nt) dacc[rt][nt].f[r] *= corr;
        }
        __builtin_amdgcn_wave_barrier();
        BFrag pf;   // P back in A-layout via per-wave LDS round-trip
        {
          ((uint4*)pf.h)[0] = *(const uint4*)&Pscr[wave][ln][half * 8];
          ((uint4*)pf.h)[1] = *(const uint4*)&Pscr[wave][ln][16 + half * 8];
        }
        #pragma unroll
        for (int nt = 0; nt < 4; ++nt)
          dacc[rt][nt].v = wmma_bf16(pf.v, VB[nt].v, dacc[rt][nt].v);
      }
    }
    __syncthreads();

    // finalize: O += (1-g) * softmax@V
    for (int rt = 0; rt < 4; ++rt) {
      const int r0 = srow0 + rt * 16;
      #pragma unroll
      for (int nt = 0; nt < 4; ++nt) {
        const int e = nt * 16 + ln;
        #pragma unroll
        for (int r = 0; r < 8; ++r) {
          const int srow = r0 + r + half * 8;
          const float dotv = dacc[rt][nt].f[r] / ssum[srow];
          const long f = (long)c * (H * S * DH) + (long)h * (S * DH) + (long)srow * DH + e;
          const long l = f & (L - 1), dd = f >> 12;
          O[((long)b * L + l) * D + dd] += (1.f - g) * dotv;
        }
      }
    }
    __syncthreads();

    // ================= Phase C: M += sk^T @ val ; z += colsum(sk) =================
    {
      const int d0 = tid >> 2, e0 = (tid & 3) * 16;
      float macc[16];
      #pragma unroll
      for (int j = 0; j < 16; ++j) macc[j] = 0.f;
      float zacc = 0.f;
      for (int sc = 0; sc < 16; ++sc) {
        __syncthreads();
        {
          const int e = tid * 8;
          const int trow = e >> 6, tcol = e & 63;
          BFrag tmp;   // 8 halfs of one sk row -> transposed scatter
          ((uint4*)tmp.h)[0] = *(const uint4*)&skb[(long)(sc * 32 + trow) * DH + tcol];
          #pragma unroll
          for (int j = 0; j < 8; ++j) SKchT[tcol + j][trow] = tmp.h[j];
          // val chunk: contiguous memory->LDS copy (async path when available)
#ifdef USE_ASYNC_LDS
          async_copy_b128(&vlb[(long)(sc * 32 + trow) * DH + tcol], &VLch[trow][tcol]);
          async_wait0();
#else
          *(uint4*)&VLch[trow][tcol] = *(const uint4*)&vlb[(long)(sc * 32 + trow) * DH + tcol];
#endif
        }
        __syncthreads();
        #pragma unroll
        for (int s0 = 0; s0 < 32; s0 += 8) {
          BFrag sk8;   // 8 sigma(k) values along s for fixed d0: contiguous
          ((uint4*)sk8.h)[0] = *(const uint4*)&SKchT[d0][s0];
          #pragma unroll
          for (int t = 0; t < 8; ++t) {
            const int s = s0 + t;
            const float skv = (float)sk8.h[t];
            if ((tid & 3) == 0) zacc += skv;
            BFrag vl;  // 16 val values along e for fixed s: contiguous
            ((uint4*)vl.h)[0] = *(const uint4*)&VLch[s][e0];
            ((uint4*)vl.h)[1] = *(const uint4*)&VLch[s][e0 + 8];
            #pragma unroll
            for (int j = 0; j < 16; ++j) macc[j] += skv * (float)vl.h[j];
          }
        }
      }
      __syncthreads();
      #pragma unroll
      for (int j = 0; j < 16; ++j) Msh[d0][e0 + j] += macc[j];
      if ((tid & 3) == 0) zsh[d0] += zacc;
      __syncthreads();
    }
  }
}

// ---------------------------------------------------------------------------
// Host launcher
// ---------------------------------------------------------------------------
extern "C" void kernel_launch(void* const* d_in, const int* in_sizes, int n_in,
                              void* d_out, int out_size, void* d_ws, size_t ws_size,
                              hipStream_t stream) {
  (void)in_sizes; (void)n_in; (void)out_size; (void)ws_size;
  const float* query  = (const float*)d_in[0];
  const float* key_in = (const float*)d_in[1];
  const float* value  = (const float*)d_in[2];
  const float* Wq = (const float*)d_in[3];
  const float* bq = (const float*)d_in[4];
  const float* Wk = (const float*)d_in[5];
  const float* bk = (const float*)d_in[6];
  const float* Wv = (const float*)d_in[7];
  const float* bv = (const float*)d_in[8];
  const float* Wo = (const float*)d_in[9];
  const float* bo = (const float*)d_in[10];
  const float* beta = (const float*)d_in[11];
  const int* use_mask   = (const int*)d_in[14];
  const int* delta_rule = (const int*)d_in[15];

  const long NT = 16384;   // B*L
  const int  D  = 1024;

  char* ws = (char*)d_ws;
  float*  Qf  = (float*)(ws);                     // 64 MB
  float*  Kf  = (float*)(ws + (64ll  << 20));     // 64 MB
  float*  Vf  = (float*)(ws + (128ll << 20));     // 64 MB
  float*  Of  = (float*)(ws + (192ll << 20));     // 64 MB
  __bf16* act = (__bf16*)(ws + (256ll << 20));    // 32 MB
  __bf16* wbf = (__bf16*)(ws + (288ll << 20));    //  2 MB
  __bf16* skb = (__bf16*)(ws + (292ll << 20));    //  4 MB
  __bf16* vlb = (__bf16*)(ws + (296ll << 20));    //  4 MB

  const long nact = NT * D;
  const long nw   = (long)D * D;
  const dim3 cvtB(256);
  const dim3 cvtGa((unsigned)(nact / 256)), cvtGw((unsigned)(nw / 256));
  const dim3 gemmG(D / GM_BN, (unsigned)(NT / GM_BM)), gemmB(256);

  // Q = query @ Wq + bq
  cvt_f32_bf16<<<cvtGa, cvtB, 0, stream>>>(query, act, nact);
  cvt_f32_bf16<<<cvtGw, cvtB, 0, stream>>>(Wq, wbf, nw);
  gemm_bf16_f32<<<gemmG, gemmB, 0, stream>>>(act, wbf, bq, Qf, (int)NT, D, D);
  // K = key_in @ Wk + bk
  cvt_f32_bf16<<<cvtGa, cvtB, 0, stream>>>(key_in, act, nact);
  cvt_f32_bf16<<<cvtGw, cvtB, 0, stream>>>(Wk, wbf, nw);
  gemm_bf16_f32<<<gemmG, gemmB, 0, stream>>>(act, wbf, bk, Kf, (int)NT, D, D);
  // V = value @ Wv + bv
  cvt_f32_bf16<<<cvtGa, cvtB, 0, stream>>>(value, act, nact);
  cvt_f32_bf16<<<cvtGw, cvtB, 0, stream>>>(Wv, wbf, nw);
  gemm_bf16_f32<<<gemmG, gemmB, 0, stream>>>(act, wbf, bv, Vf, (int)NT, D, D);

  // memory + local attention scan: one block per (b,h)
  infini_attn<<<dim3(64), dim3(256), 0, stream>>>(Qf, Kf, Vf, Of, skb, vlb,
                                                  beta, use_mask, delta_rule);

  // out = O @ Wo + bo
  cvt_f32_bf16<<<cvtGa, cvtB, 0, stream>>>(Of, act, nact);
  cvt_f32_bf16<<<cvtGw, cvtB, 0, stream>>>(Wo, wbf, nw);
  gemm_bf16_f32<<<gemmG, gemmB, 0, stream>>>(act, wbf, bo, (float*)d_out, (int)NT, D, D);
}
